// Attention_68693706932380
// MI455X (gfx1250) — matrix operand order repaired and checked by reading, not compiled, not run on hardware
//
#include <hip/hip_runtime.h>

#define N_ROWS 8192
#define E_DIM  512
#define DK_INV 0.044194173824159216f                 // 1/sqrt(512)
#define SCORE_SCALE 0.063758021631099357f            // DK_INV * log2(e)
#define LN_EPS 1e-6f

typedef unsigned short u16;
typedef __attribute__((ext_vector_type(16))) __bf16 bf16x16;
typedef __attribute__((ext_vector_type(8)))  __bf16 bf16x8;
typedef __attribute__((ext_vector_type(8)))  float  f32x8;

__device__ __forceinline__ u16 f32_to_bf16(float f) {
  unsigned int u = __float_as_uint(f);
  u += 0x7FFFu + ((u >> 16) & 1u);   // round-to-nearest-even
  return (u16)(u >> 16);
}
__device__ __forceinline__ float bf16_to_f32(u16 h) {
  return __uint_as_float(((unsigned int)h) << 16);
}
// Raw v_exp_f32. Args are always <= 0 here; flushing exp2(x < -126) to 0 is
// numerically irrelevant for logsumexp accumulation (sum >= 1).
__device__ __forceinline__ float fast_exp2(float x) {
  return __builtin_amdgcn_exp2f(x);
}

// ---------------------------------------------------------------- convert
__global__ void cvt_bf16_kernel(const float* __restrict__ src,
                                u16* __restrict__ dst, int n) {
  int i = (blockIdx.x * blockDim.x + threadIdx.x) * 4;
  if (i + 3 < n) {
    float4 v = *(const float4*)(src + i);
    ushort4 o;
    o.x = f32_to_bf16(v.x); o.y = f32_to_bf16(v.y);
    o.z = f32_to_bf16(v.z); o.w = f32_to_bf16(v.w);
    *(ushort4*)(dst + i) = o;
  }
}

// ---------------------------------------------------------------- path proj (tiny)
__global__ void path_proj_kernel(const float* __restrict__ W, const float* __restrict__ b,
                                 const float* __restrict__ path, float* __restrict__ outv) {
  int e = blockIdx.x * blockDim.x + threadIdx.x;
  if (e < E_DIM) {
    float a = b[e];
    for (int k = 0; k < E_DIM; ++k) a += W[(size_t)e * E_DIM + k] * path[k];
    outv[e] = a;
  }
}

// ---------------------------------------------------------------- projection GEMM
// out[n,e] = (sum_k A[n,k] * W[e,k] + bias[e]) * scale; A,W bf16.
// Block = 256 thr (8 waves), tile 128 rows x 64 cols, wave tile 32x32.
template <bool OUT_BF16>
__global__ __launch_bounds__(256)
void proj_gemm_kernel(const u16* __restrict__ Ab, const u16* __restrict__ Wb,
                      const float* __restrict__ bias, float scale,
                      u16* __restrict__ outb, float* __restrict__ outf) {
  __shared__ u16 sA[128][40];              // 32 k + 8 pad -> conflict-free, 16B rows
  const int tid  = threadIdx.x;
  const int lane = tid & 31, wave = tid >> 5;
  const int hi = lane >> 4, lm16 = lane & 15;
  const int wr = wave & 3, wc = wave >> 2;
  const int rowBlock = blockIdx.x * 128;
  const int rowBase  = rowBlock + wr * 32;
  const int colBase  = blockIdx.y * 64 + wc * 32;

  f32x8 acc[2][2];
#pragma unroll
  for (int a = 0; a < 2; ++a)
#pragma unroll
    for (int b = 0; b < 2; ++b)
#pragma unroll
      for (int r = 0; r < 8; ++r) acc[a][b][r] = 0.f;

  const int ldRow = tid >> 1, ldSeg = tid & 1;

  for (int k0 = 0; k0 < E_DIM; k0 += 32) {
    __syncthreads();
    {   // stage A tile: 128 x 32 bf16 (each thread copies 16 contiguous bf16)
      const u16* src = Ab + (size_t)(rowBlock + ldRow) * E_DIM + k0 + ldSeg * 16;
      uint4 v0 = *(const uint4*)(src);
      uint4 v1 = *(const uint4*)(src + 8);
      *(uint4*)&sA[ldRow][ldSeg * 16]     = v0;
      *(uint4*)&sA[ldRow][ldSeg * 16 + 8] = v1;
    }
    __syncthreads();

    bf16x16 afrag[2], bfrag[2];
#pragma unroll
    for (int rs = 0; rs < 2; ++rs) {
      const u16* rp = &sA[wr * 32 + rs * 16 + lm16][0];
      bf16x8 lo  = *(const bf16x8*)(rp + 16 * hi);
      bf16x8 hi8 = *(const bf16x8*)(rp + 16 * hi + 8);
      afrag[rs] = __builtin_shufflevector(lo, hi8, 0,1,2,3,4,5,6,7,8,9,10,11,12,13,14,15);
    }
#pragma unroll
    for (int cs = 0; cs < 2; ++cs) {
      const u16* bp = Wb + (size_t)(colBase + cs * 16 + lm16) * E_DIM + k0 + 16 * hi;
      bfrag[cs] = *(const bf16x16*)bp;
    }
#pragma unroll
    for (int rs = 0; rs < 2; ++rs)
#pragma unroll
      for (int cs = 0; cs < 2; ++cs)
        acc[rs][cs] = __builtin_amdgcn_wmma_f32_16x16x32_bf16(
            false, afrag[rs], false, bfrag[cs], (short)0, acc[rs][cs], false, false);
  }

#pragma unroll
  for (int rs = 0; rs < 2; ++rs)
#pragma unroll
    for (int cs = 0; cs < 2; ++cs) {
      const int col = colBase + cs * 16 + lm16;
      const float bv = bias[col];
#pragma unroll
      for (int r = 0; r < 8; ++r) {
        const int row = rowBase + rs * 16 + r + 8 * hi;
        const float v = (acc[rs][cs][r] + bv) * scale;
        if (OUT_BF16) outb[(size_t)row * E_DIM + col] = f32_to_bf16(v);
        else          outf[(size_t)row * E_DIM + col] = v;
      }
    }
}

// ---------------------------------------------------------------- attention row stats
// Q is pre-scaled by DK_INV*log2(e): scores come out of WMMA already in the
// base-2 log domain, so the online logsumexp uses bare v_exp_f32.
// 128 Q rows staged fully in dynamic LDS (130KB); K streamed from L2.
__global__ __launch_bounds__(256)
void attn_stats_kernel(const u16* __restrict__ Qb, const u16* __restrict__ Kb,
                       float* __restrict__ Mp, float* __restrict__ Sp,
                       int jTilesPerBlock) {
  extern __shared__ char smem[];
  u16 (*qs)[520] = (u16 (*)[520])smem;                       // 128 x (512+8) bf16
  float* lmax = (float*)(smem + 128 * 520 * sizeof(u16));    // [128][2]
  float* lsum = lmax + 256;                                  // [128][2]

  const int tid  = threadIdx.x;
  const int lane = tid & 31, wave = tid >> 5;
  const int hi = lane >> 4, lm16 = lane & 15;
  const int wr = wave & 3, wc = wave >> 2;
  const int rowBlock = blockIdx.x * 128;

  {   // stage 128 full Q rows once
    const int ldRow = tid >> 1, ldSeg = tid & 1;
    const u16* src = Qb + (size_t)(rowBlock + ldRow) * E_DIM + ldSeg * 256;
    u16* dst = &qs[ldRow][ldSeg * 256];
#pragma unroll
    for (int c = 0; c < 32; ++c)
      *(uint4*)(dst + c * 8) = *(const uint4*)(src + c * 8);
  }
  __syncthreads();

  float runM[16], runS[16];
#pragma unroll
  for (int s = 0; s < 16; ++s) { runM[s] = -__builtin_inff(); runS[s] = 0.f; }

  const int j0Base = blockIdx.y * jTilesPerBlock * 64;
  for (int jt = 0; jt < jTilesPerBlock; ++jt) {
    const int j0 = j0Base + jt * 64;
    f32x8 acc[2][2];
#pragma unroll
    for (int a = 0; a < 2; ++a)
#pragma unroll
      for (int b = 0; b < 2; ++b)
#pragma unroll
        for (int r = 0; r < 8; ++r) acc[a][b][r] = 0.f;

#pragma unroll 2
    for (int k0 = 0; k0 < E_DIM; k0 += 32) {
      bf16x16 afrag[2], bfrag[2];
#pragma unroll
      for (int rs = 0; rs < 2; ++rs) {
        const u16* rp = &qs[wr * 32 + rs * 16 + lm16][k0];
        bf16x8 lo  = *(const bf16x8*)(rp + 16 * hi);
        bf16x8 hi8 = *(const bf16x8*)(rp + 16 * hi + 8);
        afrag[rs] = __builtin_shufflevector(lo, hi8, 0,1,2,3,4,5,6,7,8,9,10,11,12,13,14,15);
      }
#pragma unroll
      for (int cs = 0; cs < 2; ++cs) {
        const u16* bp = Kb + (size_t)(j0 + wc * 32 + cs * 16 + lm16) * E_DIM + k0 + 16 * hi;
        if (cs == 0 && k0 + 32 < E_DIM) __builtin_prefetch((const void*)(bp + 32), 0, 1);
        bfrag[cs] = *(const bf16x16*)bp;
      }
#pragma unroll
      for (int rs = 0; rs < 2; ++rs)
#pragma unroll
        for (int cs = 0; cs < 2; ++cs)
          acc[rs][cs] = __builtin_amdgcn_wmma_f32_16x16x32_bf16(
              false, afrag[rs], false, bfrag[cs], (short)0, acc[rs][cs], false, false);
    }

    // branchless online logsumexp update, base 2, raw v_exp_f32.
    // Per lane: 16 row-slots, 2 columns each this tile (fused into one max).
#pragma unroll
    for (int rs = 0; rs < 2; ++rs)
#pragma unroll
      for (int r = 0; r < 8; ++r) {
        const int slot = rs * 8 + r;
        const float x0 = acc[rs][0][r];            // already log2-domain
        const float x1 = acc[rs][1][r];
        const float m  = runM[slot];
        const float mn = fmaxf(m, fmaxf(x0, x1));
        runS[slot] = runS[slot] * fast_exp2(m - mn)
                   + fast_exp2(x0 - mn) + fast_exp2(x1 - mn);
        runM[slot] = mn;
      }
  }

  // reduce across the 16 lanes of each half-wave that share a row
#pragma unroll
  for (int slot = 0; slot < 16; ++slot) {
    float m = runM[slot], s = runS[slot];
#pragma unroll
    for (int off = 8; off >= 1; off >>= 1) {
      float mo = __shfl_xor(m, off, 32);
      float so = __shfl_xor(s, off, 32);
      float mn = fmaxf(m, mo);
      s = s * fast_exp2(m - mn) + so * fast_exp2(mo - mn);
      m = mn;
    }
    if (lm16 == 0) {
      const int rowL = wr * 32 + (slot >> 3) * 16 + (slot & 7) + 8 * hi;
      lmax[rowL * 2 + wc] = m;
      lsum[rowL * 2 + wc] = s;
    }
  }
  __syncthreads();
  if (tid < 128) {
    const float m0 = lmax[tid * 2], m1 = lmax[tid * 2 + 1];
    const float s0 = lsum[tid * 2], s1 = lsum[tid * 2 + 1];
    const float mn = fmaxf(m0, m1);
    const size_t idx = (size_t)blockIdx.y * N_ROWS + rowBlock + tid;
    Mp[idx] = mn;
    Sp[idx] = s0 * fast_exp2(m0 - mn) + s1 * fast_exp2(m1 - mn);
  }
}

// ---------------------------------------------------------------- finalize: softmax split + residual + LayerNorm
__global__ __launch_bounds__(128)
void finalize_kernel(const float* __restrict__ env, const u16* __restrict__ Kb,
                     const float* __restrict__ pq, const float* __restrict__ pv,
                     const float* __restrict__ Vf,
                     const float* __restrict__ Mp, const float* __restrict__ Sp,
                     const float* __restrict__ gamma, const float* __restrict__ beta,
                     float* __restrict__ out, int nJ) {
  __shared__ float red[128];
  const int i = blockIdx.x, t = threadIdx.x;

  // merge j-split log2-sum-exp partials
  float m = Mp[i], s = Sp[i];
  for (int j = 1; j < nJ; ++j) {
    const float mj = Mp[(size_t)j * N_ROWS + i];
    const float sj = Sp[(size_t)j * N_ROWS + i];
    const float mn = fmaxf(m, mj);
    s = s * fast_exp2(m - mn) + sj * fast_exp2(mj - mn);
    m = mn;
  }

  // path score (log2 domain): p_i = (K[i] . path_query) * DK_INV * log2(e)
  float part = 0.f;
#pragma unroll
  for (int c = 0; c < 4; ++c) {
    const int e = t + c * 128;
    part += bf16_to_f32(Kb[(size_t)i * E_DIM + e]) * pq[e];
  }
  red[t] = part; __syncthreads();
  for (int o = 64; o > 0; o >>= 1) { if (t < o) red[t] += red[t + o]; __syncthreads(); }
  const float ps = red[0] * SCORE_SCALE;
  __syncthreads();

  const float mn = fmaxf(m, ps);
  const float eSelf = s * fast_exp2(m - mn);
  const float ePath = fast_exp2(ps - mn);
  const float denom = eSelf + ePath;
  const float aSelf = eSelf / denom, aPath = ePath / denom;

  float xv[4], lsum = 0.f;
#pragma unroll
  for (int c = 0; c < 4; ++c) {
    const int e = t + c * 128;
    const float x = env[(size_t)i * E_DIM + e] + aSelf * Vf[(size_t)i * E_DIM + e] + aPath * pv[e];
    xv[c] = x; lsum += x;
  }
  red[t] = lsum; __syncthreads();
  for (int o = 64; o > 0; o >>= 1) { if (t < o) red[t] += red[t + o]; __syncthreads(); }
  const float mu = red[0] * (1.0f / E_DIM);
  __syncthreads();
  float lvar = 0.f;
#pragma unroll
  for (int c = 0; c < 4; ++c) { const float d = xv[c] - mu; lvar += d * d; }
  red[t] = lvar; __syncthreads();
  for (int o = 64; o > 0; o >>= 1) { if (t < o) red[t] += red[t + o]; __syncthreads(); }
  const float inv = rsqrtf(red[0] * (1.0f / E_DIM) + LN_EPS);
  __syncthreads();
#pragma unroll
  for (int c = 0; c < 4; ++c) {
    const int e = t + c * 128;
    out[(size_t)i * E_DIM + e] = gamma[e] * (xv[c] - mu) * inv + beta[e];
  }
}

// ---------------------------------------------------------------- launch
extern "C" void kernel_launch(void* const* d_in, const int* in_sizes, int n_in,
                              void* d_out, int out_size, void* d_ws, size_t ws_size,
                              hipStream_t stream) {
  const float* env   = (const float*)d_in[0];
  const float* path  = (const float*)d_in[1];
  const float* Wq    = (const float*)d_in[2];
  const float* bq    = (const float*)d_in[3];
  const float* Wk    = (const float*)d_in[4];
  const float* bk    = (const float*)d_in[5];
  const float* Wv    = (const float*)d_in[6];
  const float* bv    = (const float*)d_in[7];
  const float* gamma = (const float*)d_in[8];
  const float* beta  = (const float*)d_in[9];
  float* out = (float*)d_out;

  char* ws = (char*)d_ws;
  size_t off = 0;
  auto alloc = [&](size_t bytes) -> char* {
    char* p = ws + off;
    off = (off + bytes + 255) & ~(size_t)255;
    return p;
  };
  const int NJ = 8;
  u16*   envb = (u16*)  alloc((size_t)N_ROWS * E_DIM * sizeof(u16));
  u16*   Qb   = (u16*)  alloc((size_t)N_ROWS * E_DIM * sizeof(u16));
  u16*   Kb   = (u16*)  alloc((size_t)N_ROWS * E_DIM * sizeof(u16));
  u16*   Wqb  = (u16*)  alloc((size_t)E_DIM * E_DIM * sizeof(u16));
  u16*   Wkb  = (u16*)  alloc((size_t)E_DIM * E_DIM * sizeof(u16));
  u16*   Wvb  = (u16*)  alloc((size_t)E_DIM * E_DIM * sizeof(u16));
  float* Vf   = (float*)alloc((size_t)N_ROWS * E_DIM * sizeof(float));
  float* pqv  = (float*)alloc(E_DIM * sizeof(float));
  float* pvv  = (float*)alloc(E_DIM * sizeof(float));
  float* Mp   = (float*)alloc((size_t)NJ * N_ROWS * sizeof(float));
  float* Sp   = (float*)alloc((size_t)NJ * N_ROWS * sizeof(float));

  // 1) bf16 conversions
  cvt_bf16_kernel<<<(N_ROWS * E_DIM / 4) / 256, 256, 0, stream>>>(env, envb, N_ROWS * E_DIM);
  cvt_bf16_kernel<<<(E_DIM * E_DIM / 4) / 256, 256, 0, stream>>>(Wq, Wqb, E_DIM * E_DIM);
  cvt_bf16_kernel<<<(E_DIM * E_DIM / 4) / 256, 256, 0, stream>>>(Wk, Wkb, E_DIM * E_DIM);
  cvt_bf16_kernel<<<(E_DIM * E_DIM / 4) / 256, 256, 0, stream>>>(Wv, Wvb, E_DIM * E_DIM);

  // 2) path projections (f32, tiny)
  path_proj_kernel<<<4, 128, 0, stream>>>(Wq, bq, path, pqv);
  path_proj_kernel<<<4, 128, 0, stream>>>(Wv, bv, path, pvv);

  // 3) Q/K/V projections via WMMA (Q pre-scaled into base-2 score domain)
  dim3 gGemm(N_ROWS / 128, E_DIM / 64);
  proj_gemm_kernel<true ><<<gGemm, 256, 0, stream>>>(envb, Wqb, bq, SCORE_SCALE, Qb, nullptr);
  proj_gemm_kernel<true ><<<gGemm, 256, 0, stream>>>(envb, Wkb, bk, 1.0f,        Kb, nullptr);
  proj_gemm_kernel<false><<<gGemm, 256, 0, stream>>>(envb, Wvb, bv, 1.0f,  nullptr, Vf);

  // 4) attention row stats (online log2-sum-exp over N columns), j split NJ ways
  size_t smem = 128 * 520 * sizeof(u16) + 512 * sizeof(float);
  dim3 gAttn(N_ROWS / 128, NJ);
  attn_stats_kernel<<<gAttn, 256, smem, stream>>>(Qb, Kb, Mp, Sp, (N_ROWS / 64) / NJ);

  // 5) finalize: merge partials, softmax split, residual + LayerNorm
  finalize_kernel<<<N_ROWS, 128, 0, stream>>>(env, Kb, pqv, pvv, Vf, Mp, Sp,
                                              gamma, beta, out, NJ);
}
